// MultiQueryAttentionLayerV2_66211215835325
// MI455X (gfx1250) — matrix operand order repaired and checked by reading, not compiled, not run on hardware
//
#include <hip/hip_runtime.h>
#include <hip/hip_bf16.h>

// ---------------------------------------------------------------------------
// MQA (B=2, N=2048, D=1024, H=16, K=V=64) for gfx1250 (MI455X).
// All matmuls via v_wmma_f32_16x16x32_bf16 (wave32 WMMA), flash-attention
// fusion so scores never hit HBM. Staging uses packed bf16x4 LDS stores;
// global_prefetch_b8 issued in a separate block so staging loads can clause.
// ---------------------------------------------------------------------------

typedef __bf16 bf16;
typedef __attribute__((ext_vector_type(16))) __bf16 v16bf;
typedef __attribute__((ext_vector_type(8)))  __bf16 bf16x8;
typedef __attribute__((ext_vector_type(4)))  __bf16 bf16x4;
typedef __attribute__((ext_vector_type(8)))  float  v8f;
typedef __attribute__((ext_vector_type(4)))  float  f32x4;

union FragU { v16bf v; bf16x8 h[2]; };

// A-matrix 16x32 bf16 fragment (ISA 7.12.2): lane = hf*16 + r, M = r.
// elements 0..7  -> K = hf*8 + 0..7
// elements 8..15 -> K = 16 + hf*8 + 0..7
__device__ __forceinline__ v16bf load_frag_a(const bf16* base, int ldk, int r, int hf) {
    FragU u;
    const bf16* p = base + (size_t)r * ldk;
    u.h[0] = *(const bf16x8*)(p + hf * 8);
    u.h[1] = *(const bf16x8*)(p + 16 + hf * 8);
    return u.v;
}

// B-matrix 32x16 bf16 fragment: N = r, elements 0..7 -> K = hf*8 + 0..7, 8..15 -> K = 16 + hf*8 + 0..7.
// Source tile stored as [n][k] with k contiguous (stride ldk).
__device__ __forceinline__ v16bf load_frag_b(const bf16* base, int ldk, int r, int hf) {
    FragU u;
    const bf16* p = base + (size_t)r * ldk;
    u.h[0] = *(const bf16x8*)(p + hf * 8);
    u.h[1] = *(const bf16x8*)(p + 16 + hf * 8);
    return u.v;
}

__device__ __forceinline__ v8f wmma_bf16(v16bf a, v16bf b, v8f c) {
    return __builtin_amdgcn_wmma_f32_16x16x32_bf16(
        /*neg_a=*/false, a, /*neg_b=*/false, b,
        /*c_mod=*/(short)0, c, /*reuse_a=*/false, /*reuse_b=*/false);
}

// Reductions across the 16 lanes of one wave32 half (groups [0..15],[16..31]).
__device__ __forceinline__ float rmax16(float v) {
    v = fmaxf(v, __shfl_xor(v, 1, 16));
    v = fmaxf(v, __shfl_xor(v, 2, 16));
    v = fmaxf(v, __shfl_xor(v, 4, 16));
    v = fmaxf(v, __shfl_xor(v, 8, 16));
    return v;
}
__device__ __forceinline__ float rsum16(float v) {
    v += __shfl_xor(v, 1, 16);
    v += __shfl_xor(v, 2, 16);
    v += __shfl_xor(v, 4, 16);
    v += __shfl_xor(v, 8, 16);
    return v;
}

// Stage 4 contiguous elements into LDS as one packed bf16x4 (8B ds_store).
__device__ __forceinline__ void stage4(const float* __restrict__ src, bf16* dst) {
    f32x4 f = *(const f32x4*)src;              // global_load_b128
    bf16x4 p;
    p.x = (bf16)f.x; p.y = (bf16)f.y;          // v_cvt_pk_bf16_f32 x2
    p.z = (bf16)f.z; p.w = (bf16)f.w;
    *(bf16x4*)dst = p;                         // ds_store_b64
}
__device__ __forceinline__ void stage4(const bf16* __restrict__ src, bf16* dst) {
    *(bf16x4*)dst = *(const bf16x4*)src;       // global_load_b64 + ds_store_b64
}

// ---------------------------------------------------------------------------
// C[M][Nc] = A[M][Kd] * W[Nc][Kd]^T    (NT GEMM; used for Q-proj and out-proj)
// Block: 128 threads = 4 waves, tile 64x64, K-step 32.
// ---------------------------------------------------------------------------
template <typename TIn, typename TOut>
__global__ __launch_bounds__(128)
void gemm_nt_kernel(const TIn* __restrict__ A, const float* __restrict__ W,
                    TOut* __restrict__ C, int M, int Nc, int Kd) {
    __shared__ bf16 As[64][32];   // [m][k]
    __shared__ bf16 Bs[64][32];   // [n][k]
    const int tid  = threadIdx.x;
    const int wave = tid >> 5, lane = tid & 31;
    const int hf   = lane >> 4, r = lane & 15;
    const int wm   = wave >> 1, wn = wave & 1;
    const int rowBase = blockIdx.x * 64;
    const int colBase = blockIdx.y * 64;

    // Per-thread staging coordinates (branch-free staging body).
    const int sRow = tid >> 3;               // 0..15 (+16*i)
    const int sQ   = (tid & 7) * 4;          // k-quad offset

    v8f acc[2][2] = {};

    for (int k0 = 0; k0 < Kd; k0 += 32) {
        __syncthreads();
        // Stage A tile (64x32) and W tile (64x32); keep loads clause-able.
        #pragma unroll
        for (int i = 0; i < 4; ++i) {
            int row = sRow + i * 16;
            stage4(A + (size_t)(rowBase + row) * Kd + k0 + sQ, &As[row][sQ]);
        }
        #pragma unroll
        for (int i = 0; i < 4; ++i) {
            int row = sRow + i * 16;
            stage4(W + (size_t)(colBase + row) * Kd + k0 + sQ, &Bs[row][sQ]);
        }
        // Prefetch next k-tile (uniform branch, separate from staging loads).
        if (k0 + 32 < Kd) {
            #pragma unroll
            for (int i = 0; i < 4; ++i) {
                int row = sRow + i * 16;
                __builtin_prefetch(A + (size_t)(rowBase + row) * Kd + k0 + 32 + sQ, 0, 1);
                __builtin_prefetch(W + (size_t)(colBase + row) * Kd + k0 + 32 + sQ, 0, 1);
            }
        }
        __syncthreads();
        #pragma unroll
        for (int mi = 0; mi < 2; ++mi) {
            v16bf a = load_frag_a(&As[wm * 32 + mi * 16][0], 32, r, hf);
            #pragma unroll
            for (int ni = 0; ni < 2; ++ni) {
                v16bf b = load_frag_b(&Bs[wn * 32 + ni * 16][0], 32, r, hf);
                acc[mi][ni] = wmma_bf16(a, b, acc[mi][ni]);
            }
        }
    }
    // C-layout: VGPR j -> M = j + 8*hf, N = r.
    #pragma unroll
    for (int mi = 0; mi < 2; ++mi)
        #pragma unroll
        for (int ni = 0; ni < 2; ++ni)
            #pragma unroll
            for (int j = 0; j < 8; ++j) {
                int rr = rowBase + wm * 32 + mi * 16 + hf * 8 + j;
                int cc = colBase + wn * 32 + ni * 16 + r;
                C[(size_t)rr * Nc + cc] = (TOut)acc[mi][ni][j];
            }
}

// ---------------------------------------------------------------------------
// C[M][64] = A[M][Kd] * W[Kd][64]   (NN GEMM, narrow; used for K/V proj)
// ---------------------------------------------------------------------------
__global__ __launch_bounds__(128)
void gemm_nn64_kernel(const float* __restrict__ A, const float* __restrict__ W,
                      bf16* __restrict__ C, int M, int Kd) {
    __shared__ bf16 As[64][32];   // [m][k]
    __shared__ bf16 Bs[64][32];   // [n][k]  (transposed while staging)
    const int tid  = threadIdx.x;
    const int wave = tid >> 5, lane = tid & 31;
    const int hf   = lane >> 4, r = lane & 15;
    const int wm   = wave >> 1, wn = wave & 1;
    const int rowBase = blockIdx.x * 64;

    const int sRow = tid >> 3;               // A staging: 0..15 (+16*i)
    const int sQ   = (tid & 7) * 4;
    const int wK   = tid >> 4;               // W staging: k-row 0..7 (+8*i)
    const int wN4  = (tid & 15) * 4;         // n-quad

    v8f acc[2][2] = {};

    for (int k0 = 0; k0 < Kd; k0 += 32) {
        __syncthreads();
        #pragma unroll
        for (int i = 0; i < 4; ++i) {
            int row = sRow + i * 16;
            stage4(A + (size_t)(rowBase + row) * Kd + k0 + sQ, &As[row][sQ]);
        }
        // W is [k][n]; transpose into Bs[n][k].
        #pragma unroll
        for (int i = 0; i < 4; ++i) {
            int kk = wK + i * 8;
            f32x4 f = *(const f32x4*)(W + (size_t)(k0 + kk) * 64 + wN4);
            Bs[wN4 + 0][kk] = (bf16)f.x;
            Bs[wN4 + 1][kk] = (bf16)f.y;
            Bs[wN4 + 2][kk] = (bf16)f.z;
            Bs[wN4 + 3][kk] = (bf16)f.w;
        }
        if (k0 + 32 < Kd) {
            #pragma unroll
            for (int i = 0; i < 4; ++i) {
                int row = sRow + i * 16;
                __builtin_prefetch(A + (size_t)(rowBase + row) * Kd + k0 + 32 + sQ, 0, 1);
                __builtin_prefetch(W + (size_t)(k0 + 32 + wK + i * 8) * 64 + wN4, 0, 1);
            }
        }
        __syncthreads();
        #pragma unroll
        for (int mi = 0; mi < 2; ++mi) {
            v16bf a = load_frag_a(&As[wm * 32 + mi * 16][0], 32, r, hf);
            #pragma unroll
            for (int ni = 0; ni < 2; ++ni) {
                v16bf b = load_frag_b(&Bs[wn * 32 + ni * 16][0], 32, r, hf);
                acc[mi][ni] = wmma_bf16(a, b, acc[mi][ni]);
            }
        }
    }
    #pragma unroll
    for (int mi = 0; mi < 2; ++mi)
        #pragma unroll
        for (int ni = 0; ni < 2; ++ni)
            #pragma unroll
            for (int j = 0; j < 8; ++j) {
                int rr = rowBase + wm * 32 + mi * 16 + hf * 8 + j;
                int cc = wn * 32 + ni * 16 + r;
                C[(size_t)rr * 64 + cc] = (bf16)acc[mi][ni][j];
            }
}

// ---------------------------------------------------------------------------
// Flash MQA: one block = (b, h, 64 query rows); 4 waves * 16 rows each.
// Loops over all N keys in 32-key tiles with online softmax.
// Q/K layout: [b*N + n][1024] (Q col = h*64+k) / [b*N + m][64].
// ---------------------------------------------------------------------------
__global__ __launch_bounds__(128)
void mqa_flash_kernel(const bf16* __restrict__ Q, const bf16* __restrict__ Kw,
                      const bf16* __restrict__ Vw, bf16* __restrict__ O,
                      int Bb, int Nn, int Hh) {
    const int NT = Nn / 64;
    const int nt = blockIdx.x % NT;
    const int bh = blockIdx.x / NT;
    const int h  = bh % Hh, b = bh / Hh;

    __shared__ bf16 Ks[32][64];      // [key][dim]
    __shared__ bf16 Vs[64][32];      // [vdim][key]  (transposed)
    __shared__ bf16 Ps[4][16][32];   // per-wave P scratch [row][key]

    const int tid = threadIdx.x;
    const int wave = tid >> 5, lane = tid & 31;
    const int hf = lane >> 4, r = lane & 15;

    const size_t rowQ = (size_t)(b * Nn + nt * 64 + wave * 16);
    // Q fragments are invariant over the key loop: keep in registers.
    const bf16* qp = Q + rowQ * 1024 + h * 64;
    const v16bf aq0 = load_frag_a(qp, 1024, r, hf);       // dims 0..31
    const v16bf aq1 = load_frag_a(qp + 32, 1024, r, hf);  // dims 32..63

    v8f acc[4] = {};
    float m_i[8], l_i[8];
    #pragma unroll
    for (int j = 0; j < 8; ++j) { m_i[j] = -1e30f; l_i[j] = 0.f; }

    const bf16* Kbase = Kw + (size_t)b * Nn * 64;
    const bf16* Vbase = Vw + (size_t)b * Nn * 64;

    const int sRow = tid >> 3;               // 0..15 (+16)
    const int sOff = (tid & 7) * 8;          // dim-octet offset

    for (int m0 = 0; m0 < Nn; m0 += 32) {
        __syncthreads();
        // Stage K tile 32x64 (row-major) and V tile transposed (Vs[vdim][key]).
        #pragma unroll
        for (int i = 0; i < 2; ++i) {
            int row = sRow + i * 16;
            *(bf16x8*)&Ks[row][sOff] =
                *(const bf16x8*)(Kbase + (size_t)(m0 + row) * 64 + sOff);
        }
        #pragma unroll
        for (int i = 0; i < 2; ++i) {
            int row = sRow + i * 16;
            bf16x8 vv = *(const bf16x8*)(Vbase + (size_t)(m0 + row) * 64 + sOff);
            #pragma unroll
            for (int j = 0; j < 8; ++j) Vs[sOff + j][row] = vv[j];
        }
        // Prefetch next key tile (uniform branch, after staging loads).
        if (m0 + 32 < Nn) {
            #pragma unroll
            for (int i = 0; i < 2; ++i) {
                int row = sRow + i * 16;
                __builtin_prefetch(Kbase + (size_t)(m0 + 32 + row) * 64 + sOff, 0, 1);
                __builtin_prefetch(Vbase + (size_t)(m0 + 32 + row) * 64 + sOff, 0, 1);
            }
        }
        __syncthreads();

        // S = Q(16x64) @ K^T : two 16x16 key tiles, contraction in 2 steps.
        v8f s0 = {}, s1 = {};
        s0 = wmma_bf16(aq0, load_frag_b(&Ks[0][0],   64, r, hf), s0);
        s0 = wmma_bf16(aq1, load_frag_b(&Ks[0][32],  64, r, hf), s0);
        s1 = wmma_bf16(aq0, load_frag_b(&Ks[16][0],  64, r, hf), s1);
        s1 = wmma_bf16(aq1, load_frag_b(&Ks[16][32], 64, r, hf), s1);

        // Online softmax per row (row = hf*8 + j, 16 key cols per tile).
        #pragma unroll
        for (int j = 0; j < 8; ++j) {
            float a0 = s0[j] * 0.125f;   // 1/sqrt(64)
            float a1 = s1[j] * 0.125f;
            float mn = fmaxf(m_i[j], rmax16(fmaxf(a0, a1)));
            float alpha = __expf(m_i[j] - mn);
            m_i[j] = mn;
            float p0 = __expf(a0 - mn);
            float p1 = __expf(a1 - mn);
            l_i[j] = l_i[j] * alpha + rsum16(p0 + p1);
            acc[0][j] *= alpha; acc[1][j] *= alpha;
            acc[2][j] *= alpha; acc[3][j] *= alpha;
            int mrow = hf * 8 + j;
            Ps[wave][mrow][r]      = (bf16)p0;   // keys 0..15
            Ps[wave][mrow][16 + r] = (bf16)p1;   // keys 16..31
        }

        // acc += P(16x32) @ V(32x64); 4 output column tiles.
        v16bf ap = load_frag_a(&Ps[wave][0][0], 32, r, hf);
        #pragma unroll
        for (int vt = 0; vt < 4; ++vt) {
            v16bf bv = load_frag_b(&Vs[vt * 16][0], 32, r, hf);
            acc[vt] = wmma_bf16(ap, bv, acc[vt]);
        }
    }

    // Normalize and store O[b*N+n][h*64 + v] as bf16.
    #pragma unroll
    for (int vt = 0; vt < 4; ++vt)
        #pragma unroll
        for (int j = 0; j < 8; ++j) {
            float val = acc[vt][j] / l_i[j];
            size_t row = rowQ + hf * 8 + j;
            O[row * 1024 + h * 64 + vt * 16 + r] = (bf16)val;
        }
}

// ---------------------------------------------------------------------------
extern "C" void kernel_launch(void* const* d_in, const int* in_sizes, int n_in,
                              void* d_out, int out_size, void* d_ws, size_t ws_size,
                              hipStream_t stream) {
    const float* x  = (const float*)d_in[0];   // [2,2048,1024]
    const float* qw = (const float*)d_in[1];   // [16,64,1024] -> [1024][1024]
    const float* kw = (const float*)d_in[2];   // [1024,64]
    const float* vw = (const float*)d_in[3];   // [1024,64]
    const float* ow = (const float*)d_in[4];   // [1024,16,64] -> [1024][1024]
    float* out = (float*)d_out;

    const int M = 4096;   // B*N
    char* ws = (char*)d_ws;
    bf16* q_ws = (bf16*)(ws);                                      // 8 MB
    bf16* k_ws = (bf16*)(ws + (size_t)8 * 1024 * 1024);            // 512 KB
    bf16* v_ws = (bf16*)(ws + (size_t)8 * 1024 * 1024 + 512 * 1024);
    bf16* o_ws = (bf16*)(ws + (size_t)9 * 1024 * 1024);            // 8 MB

    dim3 blk(128);
    // Q projection: [4096x1024] @ Wq^T
    gemm_nt_kernel<float, bf16><<<dim3(64, 16), blk, 0, stream>>>(
        x, qw, q_ws, M, 1024, 1024);
    // K / V projections: [4096x1024] @ W[1024x64]
    gemm_nn64_kernel<<<dim3(64), blk, 0, stream>>>(x, kw, k_ws, M, 1024);
    gemm_nn64_kernel<<<dim3(64), blk, 0, stream>>>(x, vw, v_ws, M, 1024);
    // Fused flash MQA: grid = B*H*(N/64) = 1024 blocks.
    mqa_flash_kernel<<<dim3(1024), blk, 0, stream>>>(
        q_ws, k_ws, v_ws, o_ws, 2, 2048, 16);
    // Output projection: [4096x1024] @ Wo^T -> f32 out
    gemm_nt_kernel<bf16, float><<<dim3(64, 16), blk, 0, stream>>>(
        o_ws, ow, out, M, 1024, 1024);
}
